// CustomCLIP_21157008900137
// MI455X (gfx1250) — matrix operand order repaired and checked by reading, not compiled, not run on hardware
//
#include <hip/hip_runtime.h>

#define B_  32
#define N_  197
#define D_  512
#define ND_ 51
#define NC_ 400
#define TK_ 50

typedef __attribute__((ext_vector_type(16))) __bf16 v16bf;
typedef __attribute__((ext_vector_type(8)))  float  v8f;

union ABFrag { v16bf v; uint4 q[2]; };

__device__ __forceinline__ unsigned short f2bf(float f) {
  union { float f; unsigned u; } x; x.f = f;
  unsigned r = x.u + 0x7FFFu + ((x.u >> 16) & 1u);   // round-to-nearest-even
  return (unsigned short)(r >> 16);
}

__device__ __forceinline__ float dot4(float4 a, float4 b) {
  return a.x * b.x + a.y * b.y + a.z * b.z + a.w * b.w;
}

// async 16B global -> LDS copy (ASYNCcnt-tracked, bypasses VGPRs)
__device__ __forceinline__ void async_ld16(unsigned lds_addr, const void* gaddr) {
  asm volatile("global_load_async_to_lds_b128 %0, %1, off"
               :: "v"(lds_addr), "v"(gaddr) : "memory");
}
__device__ __forceinline__ void wait_async0() {
  asm volatile("s_wait_asynccnt 0x0" ::: "memory");
}

// ---------------- small prologue kernels ----------------

// pack-convert f32 -> bf16 (pairs -> u32)
__global__ void k_cvt(const float* __restrict__ src, unsigned* __restrict__ dst, int n32) {
  int i = blockIdx.x * blockDim.x + threadIdx.x;
  if (i >= n32) return;
  float2 f = ((const float2*)src)[i];
  dst[i] = (unsigned)f2bf(f.x) | ((unsigned)f2bf(f.y) << 16);
}

__global__ void k_base(const float* __restrict__ img, const float* __restrict__ mtf,
                       float* __restrict__ out) {
  int idx = blockIdx.x * blockDim.x + threadIdx.x;
  if (idx >= B_ * NC_) return;
  int b = idx / NC_, c = idx % NC_;
  const float4* a = (const float4*)(img + b * D_);
  const float4* t = (const float4*)(mtf + c * D_);
  float s = 0.f;
#pragma unroll 4
  for (int i = 0; i < D_ / 4; ++i) s += dot4(a[i], t[i]);
  out[idx] = s;                          // initializes output with base_logits
}

__global__ void k_vlog(const float* __restrict__ mtf, const float* __restrict__ atf,
                       float* __restrict__ vlog) {
  int idx = blockIdx.x * blockDim.x + threadIdx.x;
  if (idx >= ND_ * NC_) return;
  int n = idx / NC_, c = idx % NC_;
  const float4* a = (const float4*)(mtf + c * D_);
  const float4* t = (const float4*)(atf + ((size_t)n * NC_ + c) * D_);
  float s = 0.f;
#pragma unroll 4
  for (int i = 0; i < D_ / 4; ++i) s += dot4(a[i], t[i]);
  vlog[idx] = s;
}

__global__ void k_vsoft(const float* __restrict__ vlog, float* __restrict__ vmat) {
  int c = blockIdx.x * blockDim.x + threadIdx.x;
  if (c >= NC_) return;
  float mx = -1e30f;
  for (int n = 0; n < ND_; ++n) mx = fmaxf(mx, vlog[n * NC_ + c]);
  float s = 0.f;
  for (int n = 0; n < ND_; ++n) s += __expf(vlog[n * NC_ + c] - mx);
  float inv = 1.f / s;
  for (int n = 0; n < ND_; ++n) vmat[n * NC_ + c] = __expf(vlog[n * NC_ + c] - mx) * inv;
}

// per-b: top-50 indices of sim[b,0,0,:] (ties -> lower index, matching lax.top_k),
// then w_sel = softmax(w[b, sel])
__global__ void k_sel(const float* __restrict__ img, const float* __restrict__ locf,
                      const float* __restrict__ atf, float* __restrict__ wsel) {
  __shared__ float s0[N_];
  __shared__ float w[N_];
  __shared__ int   sel[TK_];
  __shared__ float wv[TK_];
  int b = blockIdx.x, t = threadIdx.x;
  if (t < N_) {
    const float4* L  = (const float4*)(locf + ((size_t)b * N_ + t) * D_);
    const float4* T0 = (const float4*)(atf);          // all_text[0][0]
    const float4* I  = (const float4*)(img + b * D_);
    float sa = 0.f, sw = 0.f;
#pragma unroll 4
    for (int i = 0; i < D_ / 4; ++i) { float4 l = L[i]; sa += dot4(l, T0[i]); sw += dot4(l, I[i]); }
    s0[t] = sa; w[t] = sw;
  }
  __syncthreads();
  if (t == 0) {
    for (int k = 0; k < TK_; ++k) {
      float best = -1e30f; int bi = 0;
      for (int m = 0; m < N_; ++m) if (s0[m] > best) { best = s0[m]; bi = m; }
      sel[k] = bi; s0[bi] = -2e30f;
    }
    float mx = -1e30f;
    for (int k = 0; k < TK_; ++k) { wv[k] = w[sel[k]]; mx = fmaxf(mx, wv[k]); }
    float ss = 0.f;
    for (int k = 0; k < TK_; ++k) { wv[k] = __expf(wv[k] - mx); ss += wv[k]; }
    float inv = 1.f / ss;
    for (int k = 0; k < TK_; ++k) wsel[b * TK_ + k] = wv[k] * inv;
  }
}

// ---------------- main kernel: WMMA GEMM + top-50 weighted reduce ----------------

#define NTHREADS 416                     // 13 waves: wave wv owns M-tile row wv
#define KC     64
#define SPAD   260
#define SMEM_S  (N_ * SPAD * 4)          // 204880 B f32 sim tile
#define SMEM_A1 (208 * KC * 2)           // 26624 B  bf16 A stage (one buffer)
#define SMEM_B1 (256 * KC * 2)           // 32768 B  bf16 B stage (one buffer)
#define SMEM_W  256
#define SMEM_TOT (SMEM_S + 2 * SMEM_A1 + 2 * SMEM_B1 + SMEM_W)  // 323920 B <= 320 KB

template <int CBASE, int CCOLS>
__device__ __forceinline__ void stage_async(
    const unsigned short* __restrict__ Abf, const unsigned short* __restrict__ Bbf,
    unsigned sA_lds, unsigned sB_lds, int buf, int k0, int tid) {
  // A: 208 rows (pad rows read harmless in-workspace garbage, masked later)
  // 1664 x 16B transfers, 4 even iterations at 416 threads
  unsigned abase = sA_lds + buf * SMEM_A1;
  for (int idx = tid; idx < 208 * 8; idx += NTHREADS) {
    int m = idx >> 3, j = idx & 7;
    async_ld16(abase + (unsigned)(idx << 4), Abf + m * D_ + k0 + (j << 3));
  }
  unsigned bbase = sB_lds + buf * SMEM_B1;
  for (int idx = tid; idx < CCOLS * 8; idx += NTHREADS) {
    int cl = idx >> 3, j = idx & 7;
    async_ld16(bbase + (unsigned)(idx << 4), Bbf + (CBASE + cl) * D_ + k0 + (j << 3));
  }
}

template <int CBASE, int CCOLS>
__device__ __forceinline__ void chunk_compute(
    const unsigned short* __restrict__ Abf, const unsigned short* __restrict__ Bbf,
    const float* __restrict__ vmat, float* __restrict__ out,
    float* sS, unsigned short* sA, unsigned short* sB, const float* sW,
    unsigned sA_lds, unsigned sB_lds,
    int b, int n, int tid, int wv, int half, int l15) {
  constexpr int NTILES = CCOLS / 16;    // 16 or 9, compile-time

  v8f acc[NTILES];
#pragma unroll
  for (int i = 0; i < NTILES; ++i) acc[i] = (v8f){0.f,0.f,0.f,0.f,0.f,0.f,0.f,0.f};

  // prologue: stage k-chunk 0 into buffer 0
  stage_async<CBASE, CCOLS>(Abf, Bbf, sA_lds, sB_lds, 0, 0, tid);
  wait_async0();
  __syncthreads();

  for (int kc = 0; kc < D_ / KC; ++kc) {
    const int buf = kc & 1;
    // overlap: issue async DMA of next k-chunk while doing WMMA on current
    if (kc + 1 < D_ / KC)
      stage_async<CBASE, CCOLS>(Abf, Bbf, sA_lds, sB_lds, buf ^ 1, (kc + 1) * KC, tid);

    // A frag per ISA 16-bit 16x32 layout: lane half -> K+8, VGPR4-7 -> K+16
    const unsigned short* ap =
        sA + buf * (208 * KC) + (((wv << 4) + l15) * KC) + (half << 3);
    const unsigned short* bb = sB + buf * (256 * KC);
#pragma unroll
    for (int ks = 0; ks < KC; ks += 32) {
      ABFrag af;
      af.q[0] = *(const uint4*)(ap + ks);
      af.q[1] = *(const uint4*)(ap + ks + 16);
#pragma unroll
      for (int ntg = 0; ntg < NTILES; ntg += 4) {
        ABFrag bf[4];
#pragma unroll
        for (int u = 0; u < 4; ++u) {
          if (ntg + u < NTILES) {
            const unsigned short* bp =
                bb + ((((ntg + u) << 4) + l15) * KC) + (half << 3) + ks;
            bf[u].q[0] = *(const uint4*)(bp);
            bf[u].q[1] = *(const uint4*)(bp + 16);
          }
        }
#pragma unroll
        for (int u = 0; u < 4; ++u) {
          if (ntg + u < NTILES) {
            acc[ntg + u] = __builtin_amdgcn_wmma_f32_16x16x32_bf16(
                false, af.v, false, bf[u].v, (short)0, acc[ntg + u], false, false);
          }
        }
      }
    }
    wait_async0();       // own async writes to next buffer complete
    __syncthreads();     // everyone's writes visible, current buffer free
  }
  // store f32 sim tile to LDS (C/D layout: VGPR r -> row r (+8 for upper half))
  {
    const int rbase = (wv << 4) + (half << 3);
#pragma unroll
    for (int nt = 0; nt < NTILES; ++nt) {
      const int colL = (nt << 4) + l15;
#pragma unroll
      for (int r = 0; r < 8; ++r) {
        int row = rbase + r;
        if (row < N_) sS[row * SPAD + colL] = acc[nt][r];
      }
    }
  }
  __syncthreads();
  // per-column sorted top-50 (static-index bubble insertion, registers only)
  if (tid < CCOLS) {
    float a[TK_];
#pragma unroll
    for (int i = 0; i < TK_; ++i) a[i] = -1e30f;
#pragma unroll 1
    for (int m = 0; m < N_; ++m) {
      float x = sS[m * SPAD + tid];
#pragma unroll
      for (int i = 0; i < TK_; ++i) {
        float hi = fmaxf(a[i], x);
        x = fminf(a[i], x);
        a[i] = hi;
      }
    }
    float s = 0.f;
#pragma unroll
    for (int k = 0; k < TK_; ++k) s += sW[k] * a[k];
    const int c = CBASE + tid;
    atomicAdd(&out[b * NC_ + c], s * vmat[n * NC_ + c]);
  }
  __syncthreads();
}

__global__ __launch_bounds__(NTHREADS)
void k_main(const unsigned short* __restrict__ locbf, const unsigned short* __restrict__ atbf,
            const float* __restrict__ vmat, const float* __restrict__ wsel,
            float* __restrict__ out) {
  extern __shared__ unsigned char smem[];
  float*          sS = (float*)smem;
  unsigned short* sA = (unsigned short*)(smem + SMEM_S);
  unsigned short* sB = (unsigned short*)(smem + SMEM_S + 2 * SMEM_A1);
  float*          sW = (float*)(smem + SMEM_S + 2 * SMEM_A1 + 2 * SMEM_B1);
  const unsigned  sA_lds = (unsigned)(size_t)sA;   // LDS byte addresses for async DMA
  const unsigned  sB_lds = (unsigned)(size_t)sB;

  const int n    = blockIdx.x;
  const int b    = blockIdx.y;
  const int tid  = threadIdx.x;
  const int wv   = tid >> 5;
  const int lane = tid & 31;
  const int half = lane >> 4;
  const int l15  = lane & 15;

  if (tid < TK_) sW[tid] = wsel[b * TK_ + tid];

  const unsigned short* Abf = locbf + (size_t)b * N_ * D_;
  const unsigned short* Bbf = atbf + (size_t)n * NC_ * D_;

  chunk_compute<0, 256>(Abf, Bbf, vmat, out, sS, sA, sB, sW, sA_lds, sB_lds,
                        b, n, tid, wv, half, l15);
  chunk_compute<256, 144>(Abf, Bbf, vmat, out, sS, sA, sB, sW, sA_lds, sB_lds,
                          b, n, tid, wv, half, l15);
}

// ---------------- launcher ----------------

extern "C" void kernel_launch(void* const* d_in, const int* in_sizes, int n_in,
                              void* d_out, int out_size, void* d_ws, size_t ws_size,
                              hipStream_t stream) {
  (void)in_sizes; (void)n_in; (void)out_size; (void)ws_size;
  const float* img  = (const float*)d_in[0];
  const float* locf = (const float*)d_in[1];
  const float* atf  = (const float*)d_in[2];
  const float* mtf  = (const float*)d_in[3];
  float* out = (float*)d_out;

  unsigned char* ws = (unsigned char*)d_ws;
  float* vlog = (float*)ws;                         // 51*400 f32
  float* vmat = (float*)(ws + 81664);               // 51*400 f32
  float* wsel = (float*)(ws + 163328);              // 32*50  f32
  unsigned short* locbf = (unsigned short*)(ws + 169984);    // 32*197*512 bf16
  unsigned short* atbf  = (unsigned short*)(ws + 6625280);   // 51*400*512 bf16
  // total ws use: ~27.5 MB

  const int nloc32 = B_ * N_ * D_ / 2;              // 1,613,824
  const int nat32  = ND_ * NC_ * D_ / 2;            // 5,222,400
  k_cvt  <<<(nloc32 + 255) / 256, 256, 0, stream>>>(locf, (unsigned*)locbf, nloc32);
  k_cvt  <<<(nat32 + 255) / 256, 256, 0, stream>>>(atf, (unsigned*)atbf, nat32);

  k_base <<<(B_ * NC_ + 255) / 256, 256, 0, stream>>>(img, mtf, out);
  k_vlog <<<(ND_ * NC_ + 255) / 256, 256, 0, stream>>>(mtf, atf, vlog);
  k_vsoft<<<(NC_ + 255) / 256, 256, 0, stream>>>(vlog, vmat);
  k_sel  <<<B_, 256, 0, stream>>>(img, locf, atf, wsel);

  (void)hipFuncSetAttribute((const void*)k_main,
                            hipFuncAttributeMaxDynamicSharedMemorySize, SMEM_TOT);
  k_main <<<dim3(ND_, B_), NTHREADS, SMEM_TOT, stream>>>(locbf, atbf, vmat, wsel, out);
}
